// MixerBlock_73177652789925
// MI455X (gfx1250) — matrix operand order repaired and checked by reading, not compiled, not run on hardware
//
#include <hip/hip_runtime.h>

typedef __attribute__((ext_vector_type(16))) __bf16 v16bf;
typedef __attribute__((ext_vector_type(8)))  __bf16 v8bf;
typedef __attribute__((ext_vector_type(8)))  float  v8f;
typedef __attribute__((ext_vector_type(4)))  int    v4i;

#define DM    768
#define DFF   3072
#define MTOT  16384   // 8 * 2048 rows

// LDS tile: 128 rows x 32 k of bf16 (64 data bytes/row), padded to 80B stride
// so the 16-row fragment reads (stride 80B = 20 banks) hit distinct banks.
#define TROWS      128
#define ROW_BYTES  64
#define ROW_STRIDE 80
#define TILE_BYTES (TROWS * ROW_STRIDE)   // 10240 B; 4 buffers = 40 KB

// ---------------------------------------------------------------------------
// Async global->LDS copy (CDNA5 ASYNCcnt path) with safe fallback
// ---------------------------------------------------------------------------
#if __has_builtin(__builtin_amdgcn_global_load_async_to_lds_b128)
#define HAVE_ASYNC 1
#else
#define HAVE_ASYNC 0
#endif

__device__ inline void async_copy16(const void* g, void* l) {
#if HAVE_ASYNC
    __builtin_amdgcn_global_load_async_to_lds_b128(
        (__attribute__((address_space(1))) v4i*)(g),
        (__attribute__((address_space(3))) v4i*)(l), 0, 0);
#else
    *(v8bf*)l = *(const v8bf*)g;   // global_load_b128 + ds_store_b128
#endif
}

#if HAVE_ASYNC
#if __has_builtin(__builtin_amdgcn_s_wait_asynccnt)
#define WAIT_ASYNC(n) __builtin_amdgcn_s_wait_asynccnt(n)
#else
#define WAIT_ASYNC(n) asm volatile("s_wait_asynccnt %0" :: "i"(n) : "memory")
#endif
#else
#define WAIT_ASYNC(n) ((void)0)
#endif

// Stage one 128x32 bf16 k-slab into LDS: 512 16B chunks, 2 per thread.
__device__ inline void stage_tile(const __bf16* __restrict__ src, int row0,
                                  int ldk, int k0, char* lbuf, int tid) {
    {
        const int c = tid;                     // chunk 0..255
        const int row = c >> 2, off = (c & 3) * 16;
        const char* g = (const char*)(src + (size_t)(row0 + row) * ldk + k0) + off;
        async_copy16(g, lbuf + row * ROW_STRIDE + off);
    }
    {
        const int c = tid + 256;               // chunk 256..511
        const int row = c >> 2, off = (c & 3) * 16;
        const char* g = (const char*)(src + (size_t)(row0 + row) * ldk + k0) + off;
        async_copy16(g, lbuf + row * ROW_STRIDE + off);
    }
}

// ---------------------------------------------------------------------------
// WMMA fragments from LDS (bf16 16x16x32, wave32) -- layouts per ISA 7.12.2
// ---------------------------------------------------------------------------
union V16U { v16bf v; v8bf h[2]; };

// A 16x32: lane L -> row (L&15); slots 0..7 -> k = 8*(L>=16)+i ; slots 8..15 -> +16
__device__ inline v16bf lds_A_frag(const char* lbuf, int row0, int lane) {
    const char* p = lbuf + (row0 + (lane & 15)) * ROW_STRIDE + ((lane >> 4) << 4);
    V16U u;
    u.h[0] = *(const v8bf*)(p);
    u.h[1] = *(const v8bf*)(p + 32);
    return u.v;
}

// B 32x16 (W row-major NxK): lane L -> col (L&15); slots 0..7 -> k = 8*(L>=16)+i ; slots 8..15 -> +16
__device__ inline v16bf lds_B_frag(const char* lbuf, int row0, int lane) {
    const char* p = lbuf + (row0 + (lane & 15)) * ROW_STRIDE + ((lane >> 4) << 4);
    V16U u;
    u.h[0] = *(const v8bf*)(p);
    u.h[1] = *(const v8bf*)(p + 32);
    return u.v;
}

// ---------------------------------------------------------------------------
// Block mainloop: 256 threads = 8 waves (4 M-groups x 2 N-groups).
// Block tile 128(M) x 128(N); wave tile 32(M) x 64(N) = 8 accumulators.
// Double-buffered async LDS staging of A and B k-slabs.
// ---------------------------------------------------------------------------
__device__ inline void gemm_block(const __bf16* __restrict__ X,
                                  const __bf16* __restrict__ W,
                                  int K, int blk_m, int blk_n,
                                  int lane, int wrow, int wcol,
                                  v8f acc[2][4]) {
    // layout: [A0][B0][A1][B1], each TILE_BYTES
    __shared__ char lds[4 * TILE_BYTES];
    const int tid = threadIdx.x;

#pragma unroll
    for (int fm = 0; fm < 2; ++fm)
#pragma unroll
        for (int f = 0; f < 4; ++f)
#pragma unroll
            for (int i = 0; i < 8; ++i) acc[fm][f][i] = 0.0f;

    stage_tile(X, blk_m, K, 0, lds, tid);
    stage_tile(W, blk_n, K, 0, lds + TILE_BYTES, tid);

    const int nk = K / 32;
    for (int i = 0; i < nk; ++i) {
        const int cur = i & 1;
        char* curbuf = lds + cur * (2 * TILE_BYTES);
        char* nxtbuf = lds + (cur ^ 1) * (2 * TILE_BYTES);
        if (i + 1 < nk) {
            stage_tile(X, blk_m, K, (i + 1) * 32, nxtbuf, tid);
            stage_tile(W, blk_n, K, (i + 1) * 32, nxtbuf + TILE_BYTES, tid);
            WAIT_ASYNC(4);   // only the 4 next-slab instructions may remain
        } else {
            WAIT_ASYNC(0);
        }
        __syncthreads();     // all waves' current slab is resident

        v16bf a0 = lds_A_frag(curbuf, wrow,      lane);
        v16bf a1 = lds_A_frag(curbuf, wrow + 16, lane);
#pragma unroll
        for (int f = 0; f < 4; ++f) {
            v16bf b = lds_B_frag(curbuf + TILE_BYTES, wcol + 16 * f, lane);
            acc[0][f] = __builtin_amdgcn_wmma_f32_16x16x32_bf16(
                false, a0, false, b, (short)0, acc[0][f], false, false);
            acc[1][f] = __builtin_amdgcn_wmma_f32_16x16x32_bf16(
                false, a1, false, b, (short)0, acc[1][f], false, false);
        }
        __syncthreads();     // safe to overwrite this slab next iteration
    }
}

// C/D f32 16x16 layout: lane L, reg r -> m = r + 8*(L>=16), n = (L&15)

// ---------------------------------------------------------------------------
// GEMM kernels. grid = (N/128, MTOT/128), 256 threads.
// ---------------------------------------------------------------------------
__global__ void __launch_bounds__(256)
gemm_bias_bf16_out(const __bf16* __restrict__ X, const __bf16* __restrict__ W,
                   const float* __restrict__ bias, __bf16* __restrict__ out,
                   int K, int N) {
    const int lane = threadIdx.x & 31;
    const int w    = threadIdx.x >> 5;
    const int wrow = (w & 3) * 32, wcol = (w >> 2) * 64;
    const int blk_m = blockIdx.y * 128, blk_n = blockIdx.x * 128;
    v8f acc[2][4];
    gemm_block(X, W, K, blk_m, blk_n, lane, wrow, wcol, acc);
    const int mb  = blk_m + wrow + ((lane >> 4) << 3);
    const int nlo = lane & 15;
#pragma unroll
    for (int fm = 0; fm < 2; ++fm)
#pragma unroll
        for (int f = 0; f < 4; ++f) {
            const int n = blk_n + wcol + 16 * f + nlo;
            const float bn = bias[n];
#pragma unroll
            for (int r = 0; r < 8; ++r)
                out[(size_t)(mb + 16 * fm + r) * N + n] = (__bf16)(acc[fm][f][r] + bn);
        }
}

__global__ void __launch_bounds__(256)
gemm_bias_resid_f32_out(const __bf16* __restrict__ X, const __bf16* __restrict__ W,
                        const float* __restrict__ bias,
                        const float* __restrict__ resid, float* __restrict__ out,
                        int K, int N) {
    const int lane = threadIdx.x & 31;
    const int w    = threadIdx.x >> 5;
    const int wrow = (w & 3) * 32, wcol = (w >> 2) * 64;
    const int blk_m = blockIdx.y * 128, blk_n = blockIdx.x * 128;
    v8f acc[2][4];
    gemm_block(X, W, K, blk_m, blk_n, lane, wrow, wcol, acc);
    const int mb  = blk_m + wrow + ((lane >> 4) << 3);
    const int nlo = lane & 15;
#pragma unroll
    for (int fm = 0; fm < 2; ++fm)
#pragma unroll
        for (int f = 0; f < 4; ++f) {
            const int n = blk_n + wcol + 16 * f + nlo;
            const float bn = bias[n];
#pragma unroll
            for (int r = 0; r < 8; ++r) {
                const size_t idx = (size_t)(mb + 16 * fm + r) * N + n;
                out[idx] = acc[fm][f][r] + bn + resid[idx];   // in-place safe: 1:1
            }
        }
}

__global__ void __launch_bounds__(256)
gemm_bias_selu_bf16_out(const __bf16* __restrict__ X, const __bf16* __restrict__ W,
                        const float* __restrict__ bias, __bf16* __restrict__ out,
                        int K, int N) {
    const float kScale = 1.0507009873554805f;
    const float kAlpha = 1.6732632423543772f;
    const int lane = threadIdx.x & 31;
    const int w    = threadIdx.x >> 5;
    const int wrow = (w & 3) * 32, wcol = (w >> 2) * 64;
    const int blk_m = blockIdx.y * 128, blk_n = blockIdx.x * 128;
    v8f acc[2][4];
    gemm_block(X, W, K, blk_m, blk_n, lane, wrow, wcol, acc);
    const int mb  = blk_m + wrow + ((lane >> 4) << 3);
    const int nlo = lane & 15;
#pragma unroll
    for (int fm = 0; fm < 2; ++fm)
#pragma unroll
        for (int f = 0; f < 4; ++f) {
            const int n = blk_n + wcol + 16 * f + nlo;
            const float bn = bias[n];
#pragma unroll
            for (int r = 0; r < 8; ++r) {
                float v = acc[fm][f][r] + bn;
                v = (v > 0.0f) ? kScale * v : kScale * kAlpha * expm1f(v);
                out[(size_t)(mb + 16 * fm + r) * N + n] = (__bf16)v;
            }
        }
}

// ---------------------------------------------------------------------------
// LayerNorm over a 768-wide row; emits fp32 (residual path) + bf16 (GEMM A).
// ---------------------------------------------------------------------------
__global__ void __launch_bounds__(256)
ln_fused(const float* __restrict__ x, const float* __restrict__ g,
         const float* __restrict__ b, float* __restrict__ yf,
         __bf16* __restrict__ ybf) {
    __shared__ float red[256];
    const int tid = threadIdx.x;
    const size_t row = blockIdx.x;
    const float* xr = x + row * DM;

    float lx[3];
    float s = 0.0f;
#pragma unroll
    for (int j = 0; j < 3; ++j) { lx[j] = xr[tid + 256 * j]; s += lx[j]; }
    red[tid] = s;
    __syncthreads();
    for (int off = 128; off > 0; off >>= 1) {
        if (tid < off) red[tid] += red[tid + off];
        __syncthreads();
    }
    const float mu = red[0] * (1.0f / DM);
    __syncthreads();

    float s2 = 0.0f;
#pragma unroll
    for (int j = 0; j < 3; ++j) { const float d = lx[j] - mu; s2 += d * d; }
    red[tid] = s2;
    __syncthreads();
    for (int off = 128; off > 0; off >>= 1) {
        if (tid < off) red[tid] += red[tid + off];
        __syncthreads();
    }
    const float rstd = rsqrtf(red[0] * (1.0f / DM) + 1e-5f);

#pragma unroll
    for (int j = 0; j < 3; ++j) {
        const int i = tid + 256 * j;
        const float v = (lx[j] - mu) * rstd * g[i] + b[i];
        yf[row * DM + i]  = v;
        ybf[row * DM + i] = (__bf16)v;
    }
}

// ---------------------------------------------------------------------------
// Weight prep
// ---------------------------------------------------------------------------
__global__ void f32_to_bf16_kernel(const float* __restrict__ src,
                                   __bf16* __restrict__ dst, int n) {
    const int i = blockIdx.x * 256 + threadIdx.x;
    if (i < n) dst[i] = (__bf16)src[i];
}

// Gather V rows from interleaved in_weight: head h, elem e, comp c=2
// -> source row h*192 + e*3 + 2; output row j = h*64 + e.
__global__ void prep_v_weights(const float* __restrict__ in_w,
                               const float* __restrict__ in_b,
                               __bf16* __restrict__ wv, float* __restrict__ bv) {
    const int idx = blockIdx.x * 256 + threadIdx.x;   // 768*768 total
    const int j = idx / DM, d = idx % DM;
    const int src = (j >> 6) * 192 + (j & 63) * 3 + 2;
    wv[idx] = (__bf16)in_w[(size_t)src * DM + d];
    if (d == 0) bv[j] = in_b[src];
}

// ---------------------------------------------------------------------------
extern "C" void kernel_launch(void* const* d_in, const int* in_sizes, int n_in,
                              void* d_out, int out_size, void* d_ws, size_t ws_size,
                              hipStream_t stream) {
    const float* patches   = (const float*)d_in[0];
    const float* in_weight = (const float*)d_in[1];
    const float* in_bias   = (const float*)d_in[2];
    const float* out_w     = (const float*)d_in[3];
    const float* out_b     = (const float*)d_in[4];
    const float* mlp_w1    = (const float*)d_in[5];
    const float* mlp_b1    = (const float*)d_in[6];
    const float* mlp_w2    = (const float*)d_in[7];
    const float* mlp_b2    = (const float*)d_in[8];
    const float* ln1_g     = (const float*)d_in[9];
    const float* ln1_b     = (const float*)d_in[10];
    const float* ln2_g     = (const float*)d_in[11];
    const float* ln2_b     = (const float*)d_in[12];

    char* ws = (char*)d_ws;
    size_t off = 0;
    auto alloc = [&](size_t bytes) -> void* {
        void* p = ws + off;
        off += (bytes + 255) & ~(size_t)255;
        return p;
    };
    float*  xln_f    = (float*) alloc((size_t)MTOT * DM  * 4);  // LN1 out; reused in-place as x2
    __bf16* xln_bf   = (__bf16*)alloc((size_t)MTOT * DM  * 2);
    __bf16* heads_bf = (__bf16*)alloc((size_t)MTOT * DM  * 2);
    float*  x3_f     = (float*) alloc((size_t)MTOT * DM  * 4);
    __bf16* x3_bf    = (__bf16*)alloc((size_t)MTOT * DM  * 2);
    __bf16* h_bf     = (__bf16*)alloc((size_t)MTOT * DFF * 2);
    __bf16* wv_bf    = (__bf16*)alloc((size_t)DM * DM  * 2);
    float*  bv_f     = (float*) alloc((size_t)DM * 4);
    __bf16* outw_bf  = (__bf16*)alloc((size_t)DM * DM  * 2);
    __bf16* w1_bf    = (__bf16*)alloc((size_t)DFF * DM * 2);
    __bf16* w2_bf    = (__bf16*)alloc((size_t)DM * DFF * 2);

    // Weight conversion / gather
    prep_v_weights<<<(DM * DM) / 256, 256, 0, stream>>>(in_weight, in_bias, wv_bf, bv_f);
    f32_to_bf16_kernel<<<(DM * DM + 255) / 256, 256, 0, stream>>>(out_w, outw_bf, DM * DM);
    f32_to_bf16_kernel<<<(DFF * DM + 255) / 256, 256, 0, stream>>>(mlp_w1, w1_bf, DFF * DM);
    f32_to_bf16_kernel<<<(DM * DFF + 255) / 256, 256, 0, stream>>>(mlp_w2, w2_bf, DM * DFF);

    // 1) LN1
    ln_fused<<<MTOT, 256, 0, stream>>>(patches, ln1_g, ln1_b, xln_f, xln_bf);

    // 2) V projection: heads = xln @ Wv^T + bv   (Q/K/softmax are dead code)
    gemm_bias_bf16_out<<<dim3(DM / 128, MTOT / 128), 256, 0, stream>>>(
        xln_bf, wv_bf, bv_f, heads_bf, DM, DM);

    // 3) Output projection + residual (in-place into xln_f -> x2)
    gemm_bias_resid_f32_out<<<dim3(DM / 128, MTOT / 128), 256, 0, stream>>>(
        heads_bf, outw_bf, out_b, xln_f, xln_f, DM, DM);

    // 4) LN2
    ln_fused<<<MTOT, 256, 0, stream>>>(xln_f, ln2_g, ln2_b, x3_f, x3_bf);

    // 5) MLP up + SELU
    gemm_bias_selu_bf16_out<<<dim3(DFF / 128, MTOT / 128), 256, 0, stream>>>(
        x3_bf, w1_bf, mlp_b1, h_bf, DM, DFF);

    // 6) MLP down + residual -> d_out
    gemm_bias_resid_f32_out<<<dim3(DM / 128, MTOT / 128), 256, 0, stream>>>(
        h_bf, w2_bf, mlp_b2, x3_f, (float*)d_out, DFF, DM);
}
